// VeRA_40355512714085
// MI455X (gfx1250) — compile-verified, matrix operand order checked by reading, not run on hardware
//
#include <hip/hip_runtime.h>
#include <cstdint>

// ---------------------------------------------------------------------------
// Types
// ---------------------------------------------------------------------------
typedef __attribute__((ext_vector_type(16))) __bf16 v16bf;
typedef __attribute__((ext_vector_type(8)))  float  v8f;
typedef __attribute__((ext_vector_type(4)))  unsigned int v4u;

union FragU { v16bf bf; v4u u[2]; };

#define BM 128
#define BN 256
#define BK 32
#define LDT 40   // padded LDS row stride in bf16 elements (80 B: 16B-aligned, bank-skewed)

// ---------------------------------------------------------------------------
// CDNA5 async global->LDS copy (GV mode: 64-bit vaddr, saddr = off)
// ---------------------------------------------------------------------------
__device__ __forceinline__ void async_b128(uint32_t lds_off, const void* gaddr) {
  asm volatile("global_load_async_to_lds_b128 %0, %1, off"
               :: "v"(lds_off), "v"((uint64_t)(uintptr_t)gaddr)
               : "memory");
}

__device__ __forceinline__ uint32_t lds_addr(const void* p) {
  // LDS aperture: addr[31:0] is the LDS byte offset
  return (uint32_t)(uintptr_t)p;
}

// ---------------------------------------------------------------------------
// Generic NT GEMM: C[m,n] = sum_k Amat[m,k]*Bmat[n,k] (+addsrc)(+bias)
// Amat: [M,K] bf16 row-major; Bmat: [N,K] bf16 row-major.
// Block: 256 thr = 8 waves (2x4). Block tile 128x256, K-tile 32.
// Wave tile 64x64 = 4x4 WMMA subtiles -> 16 wmma : 16 ds_load_b128 per K-step.
// ---------------------------------------------------------------------------
template<bool ADD_SRC, bool ADD_BIAS, bool OUT_BF16>
__global__ __launch_bounds__(256)
void gemm_nt_wmma(const __bf16* __restrict__ Amat,
                  const __bf16* __restrict__ Bmat,
                  const float*  __restrict__ addsrc,
                  const float*  __restrict__ bias,
                  void*         __restrict__ Cout,
                  int M, int N, int K)
{
  __shared__ __bf16 smA[2][BM * LDT];   // 2 x 10 KB
  __shared__ __bf16 smB[2][BN * LDT];   // 2 x 20 KB   (total 60 KB)

  const int tid  = threadIdx.x;
  const int wid  = tid >> 5;
  const int lane = tid & 31;
  const int mblk = blockIdx.y * BM;
  const int nblk = blockIdx.x * BN;
  const int ktiles = K / BK;

  const int wm   = (wid >> 2) * 64;        // wave M offset: 0 / 64
  const int wn   = (wid & 3) * 64;         // wave N offset: 0/64/128/192
  const int half = (lane < 16) ? 0 : 1;    // K-half / M-half selector
  const int r16  = lane & 15;

  // Cooperative tile fill (16B chunks, 4 chunks per 64B row of the K-tile):
  //   A: 128 rows x 4 chunks = 512 chunks -> 2 per thread
  //   B: 256 rows x 4 chunks = 1024 chunks -> 4 per thread
  auto issue = [&](int kt, int buf) {
    const long kbase = (long)kt * BK;
#pragma unroll
    for (int r = 0; r < 2; ++r) {
      const int c = tid + r * 256;
      const int row = c >> 2, kc = c & 3;
      async_b128(lds_addr(&smA[buf][row * LDT + kc * 8]),
                 Amat + (long)(mblk + row) * K + kbase + kc * 8);
    }
#pragma unroll
    for (int r = 0; r < 4; ++r) {
      const int c = tid + r * 256;
      const int row = c >> 2, kc = c & 3;
      async_b128(lds_addr(&smB[buf][row * LDT + kc * 8]),
                 Bmat + (long)(nblk + row) * K + kbase + kc * 8);
    }
  };

  v8f acc[4][4];
#pragma unroll
  for (int i = 0; i < 4; ++i)
#pragma unroll
    for (int j = 0; j < 4; ++j)
      acc[i][j] = (v8f){0.f, 0.f, 0.f, 0.f, 0.f, 0.f, 0.f, 0.f};

  issue(0, 0);

  for (int kt = 0; kt < ktiles; ++kt) {
    const int buf = kt & 1;
    if (kt + 1 < ktiles) {
      issue(kt + 1, buf ^ 1);
      asm volatile("s_wait_asynccnt 0x6" ::: "memory");  // current tile's 6 done
    } else {
      asm volatile("s_wait_asynccnt 0x0" ::: "memory");
    }
    __syncthreads();   // all waves' async fills of `buf` visible

    // Fragment loads matching the 16-bit WMMA A/B layout:
    // lane row = r16; lanes 0-15 hold K 0-7 & 16-23, lanes 16-31 hold K 8-15 & 24-31.
    FragU b[4];
#pragma unroll
    for (int j = 0; j < 4; ++j) {
      const __bf16* p = &smB[buf][(wn + j * 16 + r16) * LDT + half * 8];
      b[j].u[0] = *(const v4u*)p;            // ds_load_b128
      b[j].u[1] = *(const v4u*)(p + 16);
    }
#pragma unroll
    for (int i = 0; i < 4; ++i) {
      FragU a;
      const __bf16* p = &smA[buf][(wm + i * 16 + r16) * LDT + half * 8];
      a.u[0] = *(const v4u*)p;
      a.u[1] = *(const v4u*)(p + 16);
#pragma unroll
      for (int j = 0; j < 4; ++j)
        acc[i][j] = __builtin_amdgcn_wmma_f32_16x16x32_bf16(
            false, a.bf, false, b[j].bf, (short)0, acc[i][j], false, false);
    }

    __syncthreads();   // tile reads done before next iter overwrites buf^1
  }

  // Epilogue. D layout: element (m,n): vgpr v -> m = v + 8*half, lane n = r16.
#pragma unroll
  for (int i = 0; i < 4; ++i) {
#pragma unroll
    for (int j = 0; j < 4; ++j) {
      const int mb = mblk + wm + i * 16 + half * 8;
      const int nb = nblk + wn + j * 16 + r16;
      float bval = ADD_BIAS ? bias[nb] : 0.f;
#pragma unroll
      for (int v = 0; v < 8; ++v) {
        float val = acc[i][j][v];
        const long idx = (long)(mb + v) * N + nb;
        if (ADD_SRC)  val += addsrc[idx];
        if (ADD_BIAS) val += bval;
        if (OUT_BF16) ((__bf16*)Cout)[idx] = (__bf16)val;
        else          ((float*) Cout)[idx] = val;
      }
    }
  }
}

// ---------------------------------------------------------------------------
// fp32 -> bf16 bulk convert (8 elems/thread, 32B in / 16B out)
// ---------------------------------------------------------------------------
__global__ __launch_bounds__(256)
void cvt_f32_bf16(const float* __restrict__ in, __bf16* __restrict__ out, long n)
{
  long i = ((long)blockIdx.x * blockDim.x + threadIdx.x) * 8;
  if (i + 8 <= n) {
    float4 f0 = *(const float4*)(in + i);
    float4 f1 = *(const float4*)(in + i + 4);
    union { __bf16 h[8]; v4u v; } o;
    o.h[0] = (__bf16)f0.x; o.h[1] = (__bf16)f0.y;
    o.h[2] = (__bf16)f0.z; o.h[3] = (__bf16)f0.w;
    o.h[4] = (__bf16)f1.x; o.h[5] = (__bf16)f1.y;
    o.h[6] = (__bf16)f1.z; o.h[7] = (__bf16)f1.w;
    *(v4u*)(out + i) = o.v;
  }
}

// ---------------------------------------------------------------------------
// AdT[i*R + r] = bf16(d[r] * A[r*IN + i])   (transpose + scale, tiny: 4 MiB)
// ---------------------------------------------------------------------------
__global__ __launch_bounds__(256)
void make_adt(const float* __restrict__ A, const float* __restrict__ d,
              __bf16* __restrict__ AdT, int IN, int R)
{
  long idx = (long)blockIdx.x * blockDim.x + threadIdx.x;
  if (idx < (long)IN * R) {
    int i = (int)(idx / R);
    int r = (int)(idx % R);
    AdT[idx] = (__bf16)(A[(long)r * IN + i] * d[r]);
  }
}

// ---------------------------------------------------------------------------
// Launch: inputs {x, base_weight, A, B, d, b} fp32; out fp32 [4,2048,4096]
// ---------------------------------------------------------------------------
extern "C" void kernel_launch(void* const* d_in, const int* in_sizes, int n_in,
                              void* d_out, int out_size, void* d_ws, size_t ws_size,
                              hipStream_t stream)
{
  (void)in_sizes; (void)n_in; (void)out_size; (void)ws_size;

  const float* x    = (const float*)d_in[0];   // [8192, 4096]
  const float* W    = (const float*)d_in[1];   // [4096, 4096]
  const float* Amat = (const float*)d_in[2];   // [256, 4096]
  const float* Bmat = (const float*)d_in[3];   // [4096, 256]
  const float* dvec = (const float*)d_in[4];   // [256]
  const float* bias = (const float*)d_in[5];   // [4096]
  float* out = (float*)d_out;

  const int M = 8192, N = 4096, K = 4096, R = 256;

  char* ws = (char*)d_ws;
  __bf16* xb   = (__bf16*)ws;                                       // 64 MiB
  __bf16* Weff = (__bf16*)(ws + (size_t)M * K * 2);                 // 32 MiB
  __bf16* Bb   = (__bf16*)(ws + (size_t)M * K * 2
                              + (size_t)N * K * 2);                 //  2 MiB
  __bf16* AdT  = (__bf16*)(ws + (size_t)M * K * 2
                              + (size_t)N * K * 2
                              + (size_t)N * R * 2);                 //  2 MiB

  // 1) bulk converts
  cvt_f32_bf16<<<(int)(((long)M * K) / 8 / 256), 256, 0, stream>>>(x, xb, (long)M * K);
  cvt_f32_bf16<<<(int)(((long)N * R) / 8 / 256), 256, 0, stream>>>(Bmat, Bb, (long)N * R);
  make_adt<<<(int)(((long)K * R + 255) / 256), 256, 0, stream>>>(Amat, dvec, AdT, K, R);

  // 2) W_eff[o,i] = sum_r Bb[o,r]*AdT[i,r] + W[o,i]  -> bf16   (4096x4096x256)
  gemm_nt_wmma<true, false, true>
      <<<dim3(N / BN, N / BM), 256, 0, stream>>>(Bb, AdT, W, nullptr, Weff, N, N, R);

  // 3) out[m,o] = sum_i xb[m,i]*Weff[o,i] + bias[o]           (8192x4096x4096)
  gemm_nt_wmma<false, true, false>
      <<<dim3(N / BN, M / BM), 256, 0, stream>>>(xb, Weff, nullptr, bias, out, M, N, K);
}
